// CNNGNN_6012954214774
// MI455X (gfx1250) — matrix operand order repaired and checked by reading, not compiled, not run on hardware
//
#include <hip/hip_runtime.h>
#include <hip/hip_bf16.h>

typedef __attribute__((ext_vector_type(16))) _Float16 v16h;
typedef __attribute__((ext_vector_type(8)))  float    v8f;

#define N_NODES 1024   // B*T
#define T_SEQ   128
#define B_SEQ   8
#define CB      1280
#define HD      256
#define EP      2176   // 2048 edges + 128 self loops
#define NHEAD   8
#define DHEAD   32

// ---------------- feats mean pool: [N,1280,49] -> [N,1280] -----------------
__global__ void feat_pool_kernel(const float* __restrict__ feats, float* __restrict__ pooled) {
    int idx = blockIdx.x * blockDim.x + threadIdx.x;
    if (idx >= N_NODES * CB) return;
    const float* p = feats + (size_t)idx * 49;
    float s = 0.f;
#pragma unroll
    for (int j = 0; j < 49; ++j) s += p[j];
    pooled[idx] = s * (1.0f / 49.0f);
}

// ===================== FAST WMMA GEMM ======================================
// Requirements: M%16==0, K%64==0, weight stride 1, 16B-aligned rows.
// C[M,N] = act(A[M,K] @ W[N,K]^T + bias); act: 0=none, 1=relu
// One wave per 16x32 tile (two 16x16 N-tiles share A fragments); K unrolled x2.
__device__ __forceinline__ void loadA_frag(const float* __restrict__ Arow, int k0, int half,
                                           v16h& a) {
    const float4* ap0 = (const float4*)(Arow + k0 + half * 16);
    const float4* ap1 = (const float4*)(Arow + k0 + half * 16 + 8);
    float4 r0 = ap0[0], r1 = ap0[1], r2 = ap1[0], r3 = ap1[1];
    a[0]=(_Float16)r0.x; a[1]=(_Float16)r0.y; a[2]=(_Float16)r0.z; a[3]=(_Float16)r0.w;
    a[4]=(_Float16)r1.x; a[5]=(_Float16)r1.y; a[6]=(_Float16)r1.z; a[7]=(_Float16)r1.w;
    a[8]=(_Float16)r2.x; a[9]=(_Float16)r2.y; a[10]=(_Float16)r2.z; a[11]=(_Float16)r2.w;
    a[12]=(_Float16)r3.x; a[13]=(_Float16)r3.y; a[14]=(_Float16)r3.z; a[15]=(_Float16)r3.w;
}
__device__ __forceinline__ void loadB_frag(const float* __restrict__ Wrow, int k0, int half,
                                           v16h& b) {
    const float4* bp = (const float4*)(Wrow + k0 + half * 16);
    float4 s0 = bp[0], s1 = bp[1], s2 = bp[2], s3 = bp[3];
    b[0]=(_Float16)s0.x; b[1]=(_Float16)s0.y; b[2]=(_Float16)s0.z; b[3]=(_Float16)s0.w;
    b[4]=(_Float16)s1.x; b[5]=(_Float16)s1.y; b[6]=(_Float16)s1.z; b[7]=(_Float16)s1.w;
    b[8]=(_Float16)s2.x; b[9]=(_Float16)s2.y; b[10]=(_Float16)s2.z; b[11]=(_Float16)s2.w;
    b[12]=(_Float16)s3.x; b[13]=(_Float16)s3.y; b[14]=(_Float16)s3.z; b[15]=(_Float16)s3.w;
}

template <bool TWO>
__device__ __forceinline__ void gemm_fast_body(const float* __restrict__ Arow,
                                               const float* __restrict__ W0,
                                               const float* __restrict__ W1,
                                               int half, int K, v8f& acc0, v8f& acc1) {
    for (int k0 = 0; k0 < K; k0 += 64) {
        v16h a0, a1, p0, p1, q0, q1;
        loadA_frag(Arow, k0,      half, a0);
        loadA_frag(Arow, k0 + 32, half, a1);
        loadB_frag(W0,   k0,      half, p0);
        loadB_frag(W0,   k0 + 32, half, p1);
        if (TWO) {
            loadB_frag(W1, k0,      half, q0);
            loadB_frag(W1, k0 + 32, half, q1);
        }
        acc0 = __builtin_amdgcn_wmma_f32_16x16x32_f16(false, a0, false, p0, (short)0, acc0, false, false);
        if (TWO)
            acc1 = __builtin_amdgcn_wmma_f32_16x16x32_f16(false, a0, false, q0, (short)0, acc1, false, false);
        acc0 = __builtin_amdgcn_wmma_f32_16x16x32_f16(false, a1, false, p1, (short)0, acc0, false, false);
        if (TWO)
            acc1 = __builtin_amdgcn_wmma_f32_16x16x32_f16(false, a1, false, q1, (short)0, acc1, false, false);
    }
}

__global__ void gemm_wmma_fast_kernel(const float* __restrict__ A, int lda,
                                      const float* __restrict__ W, int ldw,
                                      const float* __restrict__ bias,
                                      float* __restrict__ C, int ldc,
                                      int N, int K, int act) {
    const int lane = threadIdx.x & 31;
    const int half = lane >> 4;
    const int l16  = lane & 15;
    const int mt  = blockIdx.y * 16;
    const int nt0 = blockIdx.x * 32;
    const int nt1 = nt0 + 16;
    const bool two = (nt1 < N);                    // wave-uniform
    const float* Arow = A + (size_t)(mt + l16) * lda;
    const float* W0   = W + (size_t)(nt0 + l16) * ldw;
    const float* W1   = W + (size_t)((two ? nt1 : nt0) + l16) * ldw;

    v8f acc0 = {}, acc1 = {};
    if (two) gemm_fast_body<true >(Arow, W0, W1, half, K, acc0, acc1);
    else     gemm_fast_body<false>(Arow, W0, W1, half, K, acc0, acc1);

    const int nc0 = nt0 + l16;
    float bv0 = bias ? bias[nc0] : 0.f;
#pragma unroll
    for (int r = 0; r < 8; ++r) {
        int m = mt + (half << 3) + r;
        float val = acc0[r] + bv0;
        if (act == 1) val = fmaxf(val, 0.f);
        C[(size_t)m * ldc + nc0] = val;
    }
    if (two) {
        const int nc1 = nt1 + l16;
        float bv1 = bias ? bias[nc1] : 0.f;
#pragma unroll
        for (int r = 0; r < 8; ++r) {
            int m = mt + (half << 3) + r;
            float val = acc1[r] + bv1;
            if (act == 1) val = fmaxf(val, 0.f);
            C[(size_t)m * ldc + nc1] = val;
        }
    }
}

// ===================== GENERIC (guarded, branchless) WMMA GEMM =============
// W element (n,k) at W[n*ldw + k*wstride]; act: 0=none 1=relu 2=sigmoid(*mul)
__global__ void gemm_wmma_kernel(const float* __restrict__ A, int lda,
                                 const float* __restrict__ W, int ldw, int wstride,
                                 const float* __restrict__ bias,
                                 const float* __restrict__ mul,
                                 float* __restrict__ C, int ldc,
                                 int M, int N, int K, int act) {
    const int lane = threadIdx.x & 31;
    const int half = lane >> 4;
    const int l16  = lane & 15;
    const int mt = blockIdx.y * 16;
    const int nt = blockIdx.x * 16;
    const int mrow = mt + l16;
    const int ncol = nt + l16;
    const bool mok = (mrow < M);
    const bool nok = (ncol < N);
    const int mr = mok ? mrow : (M - 1);   // clamped (valid) addresses
    const int nc = nok ? ncol : (N - 1);
    const float* Arow = A + (size_t)mr * lda;
    const float* Wrow = W + (size_t)nc * ldw;

    v8f acc = {};
    for (int k0 = 0; k0 < K; k0 += 32) {
        v16h a, b;
#pragma unroll
        for (int v = 0; v < 8; ++v) {
            // A-matrix 16x32 f16 layout (ISA 7.12.2)
            int ka = k0 + ((v >> 2) << 4) + (half << 3) + ((v & 3) << 1);
            int ka0 = ka < K ? ka : (K - 1);
            int ka1 = (ka + 1) < K ? (ka + 1) : (K - 1);
            float fa0 = Arow[ka0];
            float fa1 = Arow[ka1];
            fa0 = (mok && ka     < K) ? fa0 : 0.f;
            fa1 = (mok && ka + 1 < K) ? fa1 : 0.f;
            a[2 * v]     = (_Float16)fa0;
            a[2 * v + 1] = (_Float16)fa1;
            // B-matrix 32x16 f16 layout: same k order as A
            int kb = k0 + ((v >> 2) << 4) + (half << 3) + ((v & 3) << 1);
            int kb0 = kb < K ? kb : (K - 1);
            int kb1 = (kb + 1) < K ? (kb + 1) : (K - 1);
            float fb0 = Wrow[(size_t)kb0 * wstride];
            float fb1 = Wrow[(size_t)kb1 * wstride];
            fb0 = (nok && kb     < K) ? fb0 : 0.f;
            fb1 = (nok && kb + 1 < K) ? fb1 : 0.f;
            b[2 * v]     = (_Float16)fb0;
            b[2 * v + 1] = (_Float16)fb1;
        }
        acc = __builtin_amdgcn_wmma_f32_16x16x32_f16(false, a, false, b,
                                                     (short)0, acc, false, false);
    }
    float bv = (bias && nok) ? bias[ncol] : 0.f;
#pragma unroll
    for (int r = 0; r < 8; ++r) {
        int m = mt + (half << 3) + r;
        if (m < M && nok) {
            float val = acc[r] + bv;
            if (act == 1) {
                val = fmaxf(val, 0.f);
            } else if (act == 2) {
                val = 1.0f / (1.0f + __expf(-val));
                if (mul) val *= mul[(size_t)m * ldc + ncol];
            }
            C[(size_t)m * ldc + ncol] = val;
        }
    }
}

// ---------------- GAT: es/ed = sum(z * a_{s,d}) per (node, head) -----------
__global__ void esed_kernel(const float* __restrict__ z,
                            const float* __restrict__ as_, const float* __restrict__ ad_,
                            float* __restrict__ es, float* __restrict__ ed) {
    int t = blockIdx.x * blockDim.x + threadIdx.x;
    if (t >= N_NODES * NHEAD) return;
    int node = t >> 3, h = t & 7;
    const float* zp = z + (size_t)node * HD + h * DHEAD;
    const float* ap = as_ + h * DHEAD;
    const float* bp = ad_ + h * DHEAD;
    float s = 0.f, d = 0.f;
#pragma unroll
    for (int j = 0; j < DHEAD; ++j) { float v = zp[j]; s += v * ap[j]; d += v * bp[j]; }
    es[t] = s; ed[t] = d;
}

__device__ __forceinline__ unsigned f2ord(float f) {
    unsigned u = __float_as_uint(f);
    return (u & 0x80000000u) ? ~u : (u | 0x80000000u);
}
__device__ __forceinline__ float ord2f(unsigned u) {
    return (u & 0x80000000u) ? __uint_as_float(u & 0x7fffffffu) : __uint_as_float(~u);
}

__global__ void gat_init_kernel(float* __restrict__ gout, unsigned* __restrict__ mbuf,
                                float* __restrict__ den) {
    int i = blockIdx.x * blockDim.x + threadIdx.x;
    if (i < N_NODES * HD) gout[i] = 0.f;
    if (i < N_NODES * NHEAD) { mbuf[i] = f2ord(-1e30f); den[i] = 0.f; }
}

__device__ __forceinline__ void edge_sd(const int* ei, int b, int e, int& src, int& dst) {
    if (e < 2048) { src = ei[b * 4096 + e]; dst = ei[b * 4096 + 2048 + e]; }
    else          { src = dst = e - 2048; }
}

__global__ void edge_pass1_kernel(const int* __restrict__ ei, const float* __restrict__ es,
                                  const float* __restrict__ ed, float* __restrict__ ebuf,
                                  unsigned* __restrict__ mbuf) {
    int t = blockIdx.x * blockDim.x + threadIdx.x;
    if (t >= B_SEQ * EP * NHEAD) return;
    int h = t & 7, be = t >> 3, e = be % EP, b = be / EP;
    int src, dst; edge_sd(ei, b, e, src, dst);
    float ev = es[(b * T_SEQ + src) * NHEAD + h] + ed[(b * T_SEQ + dst) * NHEAD + h];
    ev = ev > 0.f ? ev : 0.2f * ev;          // leaky_relu(0.2)
    ebuf[t] = ev;
    atomicMax(&mbuf[(b * T_SEQ + dst) * NHEAD + h], f2ord(ev));
}

__global__ void edge_pass2_kernel(const int* __restrict__ ei, const unsigned* __restrict__ mbuf,
                                  float* __restrict__ ebuf, float* __restrict__ den) {
    int t = blockIdx.x * blockDim.x + threadIdx.x;
    if (t >= B_SEQ * EP * NHEAD) return;
    int h = t & 7, be = t >> 3, e = be % EP, b = be / EP;
    int src, dst; edge_sd(ei, b, e, src, dst);
    float m = ord2f(mbuf[(b * T_SEQ + dst) * NHEAD + h]);
    float ex = __expf(ebuf[t] - m);
    ebuf[t] = ex;
    atomicAdd(&den[(b * T_SEQ + dst) * NHEAD + h], ex);
}

__global__ void edge_pass3_kernel(const int* __restrict__ ei, const float* __restrict__ ebuf,
                                  const float* __restrict__ den, const float* __restrict__ z,
                                  float* __restrict__ gout) {
    int be = blockIdx.x;               // b*EP + e
    int e = be % EP, b = be / EP;
    int c = threadIdx.x;               // 0..255
    int h = c >> 5;
    int src, dst; edge_sd(ei, b, e, src, dst);
    float alpha = ebuf[(size_t)be * NHEAD + h] /
                  (den[(b * T_SEQ + dst) * NHEAD + h] + 1e-16f);
    atomicAdd(&gout[(size_t)(b * T_SEQ + dst) * HD + c],
              alpha * z[(size_t)(b * T_SEQ + src) * HD + c]);
}

// h = relu(LN(gout + gat_b + hres))  — one wave per node, 8 ch/lane
__global__ void gat_combine_ln_kernel(const float* __restrict__ gout,
                                      const float* __restrict__ hres,
                                      const float* __restrict__ gatb,
                                      const float* __restrict__ g, const float* __restrict__ bl,
                                      float* __restrict__ out) {
    int node = blockIdx.x * (blockDim.x >> 5) + (threadIdx.x >> 5);
    int lane = threadIdx.x & 31;
    const float* po = gout + (size_t)node * HD;
    const float* ph = hres + (size_t)node * HD;
    float v[8];
    float s = 0.f, s2 = 0.f;
#pragma unroll
    for (int j = 0; j < 8; ++j) {
        int c = lane + j * 32;
        float x = po[c] + gatb[c] + ph[c];
        v[j] = x; s += x; s2 += x * x;
    }
#pragma unroll
    for (int off = 16; off > 0; off >>= 1) {
        s  += __shfl_xor(s,  off, 32);
        s2 += __shfl_xor(s2, off, 32);
    }
    float mu = s * (1.0f / HD);
    float var = s2 * (1.0f / HD) - mu * mu;
    float rs = rsqrtf(var + 1e-5f);
#pragma unroll
    for (int j = 0; j < 8; ++j) {
        int c = lane + j * 32;
        float y = (v[j] - mu) * rs * g[c] + bl[c];
        out[(size_t)node * HD + c] = fmaxf(y, 0.f);
    }
}

// ---------------- mean/max pool over T: h[8,128,256] -> pool[8,512] --------
__global__ void pool_mm_kernel(const float* __restrict__ h, float* __restrict__ pool) {
    int b = blockIdx.x, c = threadIdx.x;
    const float* p = h + (size_t)b * T_SEQ * HD + c;
    float s = 0.f, mx = -1e30f;
    for (int t = 0; t < T_SEQ; ++t) { float v = p[(size_t)t * HD]; s += v; mx = fmaxf(mx, v); }
    pool[b * 512 + c] = s * (1.0f / T_SEQ);
    pool[b * 512 + 256 + c] = mx;
}

// ---------------- BN + relu elementwise over [8,256] -----------------------
__global__ void bnrelu_kernel(const float* __restrict__ x,
                              const float* __restrict__ g, const float* __restrict__ bb,
                              const float* __restrict__ m, const float* __restrict__ v,
                              float* __restrict__ out, int ldo) {
    int i = blockIdx.x * blockDim.x + threadIdx.x;
    if (i >= B_SEQ * HD) return;
    int r = i >> 8, c = i & 255;
    float y = (x[i] - m[c]) * rsqrtf(v[c] + 1e-5f) * g[c] + bb[c];
    out[r * ldo + c] = fmaxf(y, 0.f);
}

// ---------------- 8-token MHA (8 heads, dim 64), qkv[8,1536] -> ao[8,512] --
__global__ void mha_attn_kernel(const float* __restrict__ qkv, float* __restrict__ ao) {
    int hH = blockIdx.x;
    __shared__ float sc[8][8];
    int tid = threadIdx.x;   // 64 threads
    {
        int q = tid >> 3, k = tid & 7;
        const float* qp = qkv + q * 1536 + hH * 64;
        const float* kp = qkv + k * 1536 + 512 + hH * 64;
        float s = 0.f;
#pragma unroll
        for (int d = 0; d < 64; ++d) s += qp[d] * kp[d];
        sc[q][k] = s * 0.125f;   // 1/sqrt(64)
    }
    __syncthreads();
    if (tid < 8) {
        float mx = -1e30f;
#pragma unroll
        for (int k = 0; k < 8; ++k) mx = fmaxf(mx, sc[tid][k]);
        float sum = 0.f;
#pragma unroll
        for (int k = 0; k < 8; ++k) { float ex = __expf(sc[tid][k] - mx); sc[tid][k] = ex; sum += ex; }
        float inv = 1.0f / sum;
#pragma unroll
        for (int k = 0; k < 8; ++k) sc[tid][k] *= inv;
    }
    __syncthreads();
    {
        int d = tid;
        for (int q = 0; q < 8; ++q) {
            float o = 0.f;
#pragma unroll
            for (int k = 0; k < 8; ++k) o += sc[q][k] * qkv[k * 1536 + 1024 + hH * 64 + d];
            ao[q * 512 + hH * 64 + d] = o;
        }
    }
}

// ===========================================================================
extern "C" void kernel_launch(void* const* d_in, const int* in_sizes, int n_in,
                              void* d_out, int out_size, void* d_ws, size_t ws_size,
                              hipStream_t stream) {
    (void)in_sizes; (void)n_in; (void)out_size; (void)ws_size;
    const float* feats = (const float*)d_in[0];
    const int*   ei    = (const int*)d_in[1];
    // d_in[2] batch_idx: unused (all zeros -> whole-sequence pooling)
    int i = 3;
    auto F = [&](int idx) { return (const float*)d_in[idx]; };
    const float* att_w1 = F(i++); const float* att_b1 = F(i++);
    const float* att_w2 = F(i++); const float* att_b2 = F(i++);
    const float* proj_w1 = F(i++); const float* proj_b1 = F(i++);
    const float* proj_w2 = F(i++); const float* proj_b2 = F(i++);
    const float* in_w = F(i++); const float* in_b = F(i++);
    const float* gat_w = F(i++); const float* gat_as = F(i++);
    const float* gat_ad = F(i++); const float* gat_b = F(i++);
    const float* res_w = F(i++); const float* res_b = F(i++);
    const float* ln_g = F(i++); const float* ln_b = F(i++);
    const float* tp_w = F(i++); const float* tp_b = F(i++);
    const float* c1_w = F(i++); const float* c1_b = F(i++);
    const float* bn1_g = F(i++); const float* bn1_b = F(i++);
    const float* bn1_m = F(i++); const float* bn1_v = F(i++);
    const float* c2_w = F(i++); const float* c2_b = F(i++);
    const float* bn2_g = F(i++); const float* bn2_b = F(i++);
    const float* bn2_m = F(i++); const float* bn2_v = F(i++);
    const float* mha_in_w = F(i++); const float* mha_in_b = F(i++);
    const float* mha_out_w = F(i++); const float* mha_out_b = F(i++);
    const float* cls_w1 = F(i++); const float* cls_b1 = F(i++);
    const float* cls_w2 = F(i++); const float* cls_b2 = F(i++);
    const float* cls_w3 = F(i++); const float* cls_b3 = F(i++);

    // ---- workspace layout (floats, 256B aligned chunks) ----
    float* wsf = (float*)d_ws;
    size_t off = 0;
    auto take = [&](size_t n) { float* p = wsf + off; off += (n + 63) & ~(size_t)63; return p; };
    float* pooled = take((size_t)N_NODES * CB);
    float* a1     = take((size_t)N_NODES * 80);
    float* pa     = take((size_t)N_NODES * CB);
    float* x1     = take((size_t)N_NODES * 1024);
    float* x2     = take((size_t)N_NODES * 512);
    float* hA     = take((size_t)N_NODES * HD);
    float* hB     = take((size_t)N_NODES * HD);
    float* zb     = take((size_t)N_NODES * HD);
    float* esb    = take(N_NODES * NHEAD);
    float* edb    = take(N_NODES * NHEAD);
    float* mb_f   = take(N_NODES * NHEAD);           // order-mapped uint max buffer
    float* denb   = take(N_NODES * NHEAD);
    float* ebuf   = take((size_t)B_SEQ * EP * NHEAD);
    float* gout   = take((size_t)N_NODES * HD);
    float* hres   = take((size_t)N_NODES * HD);
    float* poolb  = take(B_SEQ * 512);
    float* comb   = take(B_SEQ * 512);
    float* c1pre  = take(B_SEQ * HD);
    float* c1o    = take(B_SEQ * HD);
    float* c2pre  = take(B_SEQ * HD);
    float* qkv    = take(B_SEQ * 1536);
    float* ao     = take(B_SEQ * 512);
    float* ao2    = take(B_SEQ * 512);
    float* z1     = take(B_SEQ * 256);
    float* z2     = take(B_SEQ * 128);

    // fast path: M%16==0, K%64==0, wstride==1 (N may be any multiple of 16)
    auto gemm_fast = [&](const float* A, int lda, const float* W, int ldw,
                         const float* bias, float* C, int ldc,
                         int M, int N, int K, int act) {
        dim3 grid((N + 31) / 32, M / 16);
        gemm_wmma_fast_kernel<<<grid, 32, 0, stream>>>(A, lda, W, ldw, bias, C, ldc, N, K, act);
    };
    auto gemm = [&](const float* A, int lda, const float* W, int ldw, int wstride,
                    const float* bias, const float* mul, float* C, int ldc,
                    int M, int N, int K, int act) {
        dim3 grid((N + 15) / 16, (M + 15) / 16);
        gemm_wmma_kernel<<<grid, 32, 0, stream>>>(A, lda, W, ldw, wstride, bias, mul,
                                                  C, ldc, M, N, K, act);
    };

    // ---- frontend ----
    feat_pool_kernel<<<(N_NODES * CB + 255) / 256, 256, 0, stream>>>(feats, pooled);
    gemm_fast(pooled, CB, att_w1, CB, att_b1, a1, 80, N_NODES, 80, CB, 1);                 // relu
    gemm(a1, 80, att_w2, 80, 1, att_b2, pooled, pa, CB, N_NODES, CB, 80, 2);               // sigmoid*pooled (K=80)
    gemm_fast(pa, CB, proj_w1, CB, proj_b1, x1, 1024, N_NODES, 1024, CB, 1);               // relu
    gemm_fast(x1, 1024, proj_w2, 1024, proj_b2, x2, 512, N_NODES, 512, 1024, 1);           // relu
    gemm_fast(x2, 512, in_w, 512, in_b, hA, HD, N_NODES, HD, 512, 0);                      // -> h

    // ---- 3 GAT layers ----
    float* hcur = hA; float* hnxt = hB;
    const int totEH = B_SEQ * EP * NHEAD;
    for (int l = 0; l < 3; ++l) {
        const float* Wl  = gat_w + (size_t)l * HD * HD;
        const float* asl = gat_as + (size_t)l * NHEAD * DHEAD;
        const float* adl = gat_ad + (size_t)l * NHEAD * DHEAD;
        const float* gbl = gat_b + (size_t)l * HD;
        const float* rwl = res_w + (size_t)l * HD * HD;
        const float* rbl = res_b + (size_t)l * HD;
        const float* lgl = ln_g + (size_t)l * HD;
        const float* lbl = ln_b + (size_t)l * HD;

        gemm_fast(hcur, HD, Wl, HD, nullptr, zb, HD, N_NODES, HD, HD, 0);                  // z
        esed_kernel<<<(N_NODES * NHEAD + 255) / 256, 256, 0, stream>>>(zb, asl, adl, esb, edb);
        gat_init_kernel<<<(N_NODES * HD + 255) / 256, 256, 0, stream>>>(gout, (unsigned*)mb_f, denb);
        edge_pass1_kernel<<<(totEH + 255) / 256, 256, 0, stream>>>(ei, esb, edb, ebuf, (unsigned*)mb_f);
        edge_pass2_kernel<<<(totEH + 255) / 256, 256, 0, stream>>>(ei, (const unsigned*)mb_f, ebuf, denb);
        edge_pass3_kernel<<<B_SEQ * EP, HD, 0, stream>>>(ei, ebuf, denb, zb, gout);
        gemm_fast(hcur, HD, rwl, HD, rbl, hres, HD, N_NODES, HD, HD, 0);                   // residual
        gat_combine_ln_kernel<<<N_NODES / 8, 256, 0, stream>>>(gout, hres, gbl, lgl, lbl, hnxt);
        float* tmp = hcur; hcur = hnxt; hnxt = tmp;
    }

    // ---- head (M=8 -> guarded path; conv taps are strided weights) ----
    pool_mm_kernel<<<B_SEQ, HD, 0, stream>>>(hcur, poolb);
    gemm(poolb, 512, tp_w, 512, 1, tp_b, nullptr, comb, 512, B_SEQ, HD, 512, 1);           // tf -> comb[:, :256]
    gemm(comb, 512, c1_w + 1, HD * 3, 3, c1_b, nullptr, c1pre, HD, B_SEQ, HD, HD, 0);      // conv1 center tap
    bnrelu_kernel<<<(B_SEQ * HD + 255) / 256, 256, 0, stream>>>(c1pre, bn1_g, bn1_b, bn1_m, bn1_v, c1o, HD);
    gemm(c1o, HD, c2_w + 2, HD * 5, 5, c2_b, nullptr, c2pre, HD, B_SEQ, HD, HD, 0);        // conv2 center tap
    bnrelu_kernel<<<(B_SEQ * HD + 255) / 256, 256, 0, stream>>>(c2pre, bn2_g, bn2_b, bn2_m, bn2_v, comb + 256, 512);
    gemm(comb, 512, mha_in_w, 512, 1, mha_in_b, nullptr, qkv, 1536, B_SEQ, 1536, 512, 0);  // qkv
    mha_attn_kernel<<<NHEAD, 64, 0, stream>>>(qkv, ao);
    gemm(ao, 512, mha_out_w, 512, 1, mha_out_b, nullptr, ao2, 512, B_SEQ, 512, 512, 0);
    gemm(ao2, 512, cls_w1, 512, 1, cls_b1, nullptr, z1, 256, B_SEQ, 256, 512, 1);          // relu
    gemm(z1, 256, cls_w2, 256, 1, cls_b2, nullptr, z2, 128, B_SEQ, 128, 256, 1);           // relu
    gemm(z2, 128, cls_w3, 128, 1, cls_b3, nullptr, (float*)d_out, 10, B_SEQ, 10, 128, 0);  // logits
}
